// VectorQuantizer_34385508172264
// MI455X (gfx1250) — matrix operand order repaired and checked by reading, not compiled, not run on hardware
//
#include <hip/hip_runtime.h>

typedef __bf16 bf16;
typedef __attribute__((ext_vector_type(16))) __bf16 bf16x16;
typedef __attribute__((ext_vector_type(8)))  float  f32x8;

#define N_VEC   16384      // 16 * 32 * 32 feature vectors
#define DIM     256        // embedding dim
#define K_CODE  8192       // codebook size
#define KSLICES 4          // k-range split per n-tile
#define KT_PER_SLICE 128   // (8192/4)/16 k-tiles per slice

// ---------------------------------------------------------------------------
// Deterministic block-wide sum with broadcast (256 threads = 8 waves, wave32)
// ---------------------------------------------------------------------------
__device__ __forceinline__ float block_reduce_sum_bcast(float v) {
  __shared__ float s_part[8];
  __shared__ float s_tot;
  const int lane = threadIdx.x & 31;
  const int w    = threadIdx.x >> 5;
  #pragma unroll
  for (int off = 16; off > 0; off >>= 1) v += __shfl_xor(v, off, 32);
  if (lane == 0) s_part[w] = v;
  __syncthreads();
  if (threadIdx.x == 0) {
    float t = 0.f;
    #pragma unroll
    for (int i = 0; i < 8; ++i) t += s_part[i];
    s_tot = t;
  }
  __syncthreads();
  return s_tot;
}

// ---------------------------------------------------------------------------
// Normalize codebook rows, split into bf16 hi/lo. One block per code row.
// ---------------------------------------------------------------------------
__global__ void __launch_bounds__(256)
vq_norm_codebook(const float* __restrict__ w,
                 bf16* __restrict__ nc_hi, bf16* __restrict__ nc_lo) {
  const int k = blockIdx.x;
  const int d = threadIdx.x;
  const float x  = w[(size_t)k * DIM + d];
  const float ss = block_reduce_sum_bcast(x * x);
  const float nv = x / fmaxf(sqrtf(ss), 1e-12f);
  const bf16 hi  = (bf16)nv;
  nc_hi[(size_t)k * DIM + d] = hi;
  nc_lo[(size_t)k * DIM + d] = (bf16)(nv - (float)hi);
}

// ---------------------------------------------------------------------------
// Normalize z vectors (gathering the channel-major transpose), split hi/lo.
// One block per feature vector n = (b,h,w); thread = channel.
// z flat index for [b][c][h][w]: b*262144 + c*1024 + (h*32+w)
// ---------------------------------------------------------------------------
__global__ void __launch_bounds__(256)
vq_norm_z(const float* __restrict__ z,
          bf16* __restrict__ nz_hi, bf16* __restrict__ nz_lo) {
  const int n  = blockIdx.x;
  const int c  = threadIdx.x;
  const int b  = n >> 10;
  const int hw = n & 1023;
  const float x  = z[(size_t)b * 262144 + (size_t)c * 1024 + hw];
  const float ss = block_reduce_sum_bcast(x * x);
  const float nv = x / fmaxf(sqrtf(ss), 1e-12f);
  const bf16 hi  = (bf16)nv;
  nz_hi[(size_t)n * DIM + c] = hi;
  nz_lo[(size_t)n * DIM + c] = (bf16)(nv - (float)hi);
}

// ---------------------------------------------------------------------------
// Fused GEMM + argmax. Block = 8 waves, each wave owns one 16-row n-tile;
// all 8 waves share LDS-staged (double-buffered) codebook k-tiles.
// bf16 hi/lo split: dist ~= a_hi*b_hi + a_hi*b_lo + a_lo*b_hi (fp32-accurate).
// THREE independent accumulators (one per product term) break the WMMA RAW
// chain 3-ways so the SIMD can overlap matrix ops instead of stalling on
// TRANS32 dependencies.
// grid = 128 n-tile groups * 4 k-slices = 512 blocks.
// ---------------------------------------------------------------------------
__global__ void __launch_bounds__(256)
vq_dist_argmax(const bf16* __restrict__ nz_hi, const bf16* __restrict__ nz_lo,
               const bf16* __restrict__ nc_hi, const bf16* __restrict__ nc_lo,
               float* __restrict__ pval, int* __restrict__ pidx) {
  __shared__ uint4 smem[2048];                 // 2 buffers x (hi 8KB | lo 8KB)
  const int tid    = threadIdx.x;
  const int lane   = tid & 31;
  const int wv     = tid >> 5;
  const int l15    = lane & 15;
  const int hl     = lane >> 4;                // half-wave select
  const int group  = blockIdx.x >> 2;          // 0..127
  const int kslice = blockIdx.x & 3;           // 0..3
  const int ntile  = group * 8 + wv;           // 0..1023

  // ---- preload A fragments: this wave's 16 z-rows over full D (resident) ----
  // A k order used here (B matches): half 0 holds K 0..15 contiguous, half 1 K 16..31
  const bf16* pa_h = nz_hi + (size_t)(ntile * 16 + l15) * DIM + hl * 16;
  const bf16* pa_l = nz_lo + (size_t)(ntile * 16 + l15) * DIM + hl * 16;
  bf16x16 a_hi[8], a_lo[8];
  #pragma unroll
  for (int t = 0; t < 8; ++t) {
    union { bf16x16 v; uint4 q[2]; } uh, ul;
    uh.q[0] = *(const uint4*)(pa_h + t * 32);
    uh.q[1] = *(const uint4*)(pa_h + t * 32 + 8);
    ul.q[0] = *(const uint4*)(pa_l + t * 32);
    ul.q[1] = *(const uint4*)(pa_l + t * 32 + 8);
    a_hi[t] = uh.v; a_lo[t] = ul.v;
  }

  float best[8]; int bidx[8];
  #pragma unroll
  for (int r = 0; r < 8; ++r) { best[r] = -3.0e38f; bidx[r] = 0; }

  // ---- stage k-tile 0 into LDS buffer 0 (one tile = 16 codes * 256 = 8KB/half)
  const size_t sbase = (size_t)(kslice * KT_PER_SLICE) * 16 * DIM;
  {
    const uint4* gh = (const uint4*)(nc_hi + sbase);
    const uint4* gl = (const uint4*)(nc_lo + sbase);
    smem[tid]       = gh[tid];
    smem[tid + 256] = gh[tid + 256];
    smem[tid + 512] = gl[tid];
    smem[tid + 768] = gl[tid + 256];
  }
  __syncthreads();

  for (int kt = 0; kt < KT_PER_SLICE; ++kt) {
    const int bsel = kt & 1;
    // prefetch next k-tile into registers (overlaps with WMMA below)
    uint4 r0, r1, r2, r3;
    const bool more = (kt + 1) < KT_PER_SLICE;
    if (more) {
      const size_t nb = sbase + (size_t)(kt + 1) * 16 * DIM;
      const uint4* gh = (const uint4*)(nc_hi + nb);
      const uint4* gl = (const uint4*)(nc_lo + nb);
      r0 = gh[tid]; r1 = gh[tid + 256];
      r2 = gl[tid]; r3 = gl[tid + 256];
    }
    // B k order used here (same as A): lane = code column n=l15,
    // half 0 holds K 0..15 contiguous, half 1 holds K 16..31.
    const bf16* lb_h = (const bf16*)(smem + bsel * 1024) + l15 * DIM + hl * 16;
    const bf16* lb_l = lb_h + 16 * DIM;      // lo half sits 8KB after hi
    f32x8 c_hh = {};                         // a_hi * b_hi
    f32x8 c_hl = {};                         // a_hi * b_lo
    f32x8 c_lh = {};                         // a_lo * b_hi
    #pragma unroll
    for (int t = 0; t < 8; ++t) {
      const bf16x16 bh = *(const bf16x16*)(lb_h + t * 32);
      const bf16x16 bl = *(const bf16x16*)(lb_l + t * 32);
      c_hh = __builtin_amdgcn_wmma_f32_16x16x32_bf16(false, a_hi[t], false, bh,
                                                     (short)0, c_hh, false, false);
      c_hl = __builtin_amdgcn_wmma_f32_16x16x32_bf16(false, a_hi[t], false, bl,
                                                     (short)0, c_hl, false, false);
      c_lh = __builtin_amdgcn_wmma_f32_16x16x32_bf16(false, a_lo[t], false, bh,
                                                     (short)0, c_lh, false, false);
    }
    // fold this 16x16 similarity tile into the running per-row argmax.
    // C layout: lane holds column n=l15; VGPR r -> row m = r + hl*8.
    const int code = (kslice * KT_PER_SLICE + kt) * 16 + l15;
    #pragma unroll
    for (int r = 0; r < 8; ++r) {
      const float v = c_hh[r] + (c_hl[r] + c_lh[r]);
      if (v > best[r]) { best[r] = v; bidx[r] = code; }
    }
    __syncthreads();                       // all waves done reading buffer bsel
    if (more) {
      uint4* s = smem + (1 - bsel) * 1024;
      s[tid] = r0; s[tid + 256] = r1; s[tid + 512] = r2; s[tid + 768] = r3;
    }
    __syncthreads();                       // commit visible before next read
  }

  // ---- cross-lane argmax over the 16 code columns (first-index tie-break) ----
  #pragma unroll
  for (int r = 0; r < 8; ++r) {
    float v = best[r]; int i = bidx[r];
    #pragma unroll
    for (int off = 8; off > 0; off >>= 1) {
      const float ov = __shfl_xor(v, off, 32);
      const int   oi = __shfl_xor(i, off, 32);
      if (ov > v || (ov == v && oi < i)) { v = ov; i = oi; }
    }
    if (l15 == 0) {
      const int row = ntile * 16 + hl * 8 + r;
      pval[row * KSLICES + kslice] = v;
      pidx[row * KSLICES + kslice] = i;
    }
  }
}

// ---------------------------------------------------------------------------
// Merge k-slice partials, gather codebook row, write out [b,c,h,w], and
// produce per-block squared-error partial sums (deterministic loss path).
// ---------------------------------------------------------------------------
__global__ void __launch_bounds__(256)
vq_gather_loss(const float* __restrict__ z, const float* __restrict__ w,
               const float* __restrict__ pval, const int* __restrict__ pidx,
               float* __restrict__ out, float* __restrict__ pblk) {
  const int n = blockIdx.x;
  const int c = threadIdx.x;
  float bv = -3.0e38f; int bi = 0;
  #pragma unroll
  for (int s2 = 0; s2 < KSLICES; ++s2) {
    const float v = pval[n * KSLICES + s2];
    const int   i = pidx[n * KSLICES + s2];
    if (v > bv || (v == bv && i < bi)) { bv = v; bi = i; }
  }
  const float q  = w[(size_t)bi * DIM + c];       // raw (unnormalized) codeword
  const int   b  = n >> 10;
  const int   hw = n & 1023;
  const size_t zi = (size_t)b * 262144 + (size_t)c * 1024 + hw;
  const float zv = z[zi];
  out[zi] = q;                                    // STE output == z_q numerically
  const float d = q - zv;
  const float s = block_reduce_sum_bcast(d * d);
  if (threadIdx.x == 0) pblk[n] = s;
}

// ---------------------------------------------------------------------------
// Final deterministic loss reduction: loss = 1.25 * mean(sq err)
// ---------------------------------------------------------------------------
__global__ void __launch_bounds__(256)
vq_loss_reduce(const float* __restrict__ pblk, float* __restrict__ loss) {
  float s = 0.f;
  for (int i = threadIdx.x; i < N_VEC; i += 256) s += pblk[i];
  s = block_reduce_sum_bcast(s);
  if (threadIdx.x == 0) *loss = s * (1.25f / 4194304.0f);
}

// ---------------------------------------------------------------------------
extern "C" void kernel_launch(void* const* d_in, const int* in_sizes, int n_in,
                              void* d_out, int out_size, void* d_ws, size_t ws_size,
                              hipStream_t stream) {
  (void)in_sizes; (void)n_in; (void)out_size; (void)ws_size;
  const float* z = (const float*)d_in[0];       // [16,256,32,32]
  const float* w = (const float*)d_in[1];       // [8192,256]
  float* out  = (float*)d_out;                  // 4,194,304 floats
  float* loss = out + (size_t)N_VEC * DIM;      // + 1 float

  // workspace carve-up (~25.7 MB)
  bf16*  nz_hi = (bf16*)d_ws;                           // 16384*256
  bf16*  nz_lo = nz_hi + (size_t)N_VEC * DIM;           // 16384*256
  bf16*  nc_hi = nz_lo + (size_t)N_VEC * DIM;           // 8192*256
  bf16*  nc_lo = nc_hi + (size_t)K_CODE * DIM;          // 8192*256
  float* pval  = (float*)(nc_lo + (size_t)K_CODE * DIM);// 16384*4
  int*   pidx  = (int*)(pval + (size_t)N_VEC * KSLICES);// 16384*4
  float* pblk  = (float*)(pidx + (size_t)N_VEC * KSLICES); // 16384

  vq_norm_codebook<<<K_CODE, 256, 0, stream>>>(w, nc_hi, nc_lo);
  vq_norm_z       <<<N_VEC,  256, 0, stream>>>(z, nz_hi, nz_lo);
  vq_dist_argmax  <<<512,    256, 0, stream>>>(nz_hi, nz_lo, nc_hi, nc_lo, pval, pidx);
  vq_gather_loss  <<<N_VEC,  256, 0, stream>>>(z, w, pval, pidx, out, pblk);
  vq_loss_reduce  <<<1,      256, 0, stream>>>(pblk, loss);
}
